// RSSM_53781580481114
// MI455X (gfx1250) — matrix operand order repaired and checked by reading, not compiled, not run on hardware
//
#include <hip/hip_runtime.h>

// ---------------------------------------------------------------------------
// RSSM on gfx1250: bf16 WMMA (v_wmma_f32_16x16x32_bf16) for all GEMMs, fp32
// accumulate. All GEMM operands bf16 (weights converted once; activations
// emitted bf16 by producer kernels). One wave owns a full M=32 x N=16 output
// strip: each B (weight) fragment is loaded once and used by 2 WMMAs, halving
// weight traffic and doubling WMMA density per byte.
// ---------------------------------------------------------------------------

typedef __bf16 bf16x16 __attribute__((ext_vector_type(16)));
typedef float  f32x8   __attribute__((ext_vector_type(8)));

#define DEVI __device__ __forceinline__

// Problem dims
constexpr int B_ = 32, S_ = 512, E_ = 1024, A_ = 2, H_ = 1024, Z_ = 512, L_ = 64;
constexpr int CQ_ = H_ + E_ + L_;   // 2112
constexpr int CP_ = H_ + L_;        // 1088

// Output offsets (floats): (h_seq, sp, mp, gp, h_seq, sq, mq, gq)
constexpr size_t O0 = 0;
constexpr size_t O1 = O0 + (size_t)B_ * S_ * H_;
constexpr size_t O2 = O1 + (size_t)B_ * S_ * Z_;
constexpr size_t O3 = O2 + (size_t)B_ * S_ * Z_;
constexpr size_t O4 = O3 + (size_t)B_ * S_ * Z_;
constexpr size_t O5 = O4 + (size_t)B_ * S_ * H_;
constexpr size_t O6 = O5 + (size_t)B_ * S_ * Z_;
constexpr size_t O7 = O6 + (size_t)B_ * S_ * Z_;

DEVI f32x8 wmma_bf16(bf16x16 a, bf16x16 b, f32x8 c) {
  return __builtin_amdgcn_wmma_f32_16x16x32_bf16(false, a, false, b, (short)0, c,
                                                 false, false);
}

DEVI float lrelu(float x) { return x >= 0.f ? x : 0.01f * x; }
DEVI float sigm(float x)  { return 1.f / (1.f + __expf(-x)); }
DEVI float tanh_(float x) {
  x = fminf(fmaxf(x, -15.f), 15.f);
  float e = __expf(-2.f * x);
  return (1.f - e) / (1.f + e);
}

// A fragment (16x32) from bf16 source. rowBase points at row of the m-tile;
// stride in elements. ISA layout: lane L -> row L&15; lanes<16: K {0..7,16..23},
// lanes>=16: K {8..15,24..31}. Two 16B loads per lane.
DEVI bf16x16 a_frag(const __bf16* rowBase, size_t stride, int kOff, int lane) {
  const __bf16* p = rowBase + (size_t)(lane & 15) * stride + kOff + ((lane >> 4) << 3);
  union { uint4 u[2]; bf16x16 v; } c;
  c.u[0] = *(const uint4*)(p);        // K: kOff + half*8 .. +7
  c.u[1] = *(const uint4*)(p + 16);   // K: +16
  return c.v;
}

// B fragment (32x16) from row-major W[N][K] (out = x @ W^T -> B[k][n]=W[n][k]).
// colBase points at row nBase of W; ld = K. Lane L -> col L&15; lanes<16 hold
// K {0..7,16..23}, lanes>=16 hold K {8..15,24..31} (same order as a_frag).
DEVI bf16x16 b_frag(const __bf16* colBase, size_t ld, int kOff, int lane) {
  const __bf16* p = colBase + (size_t)(lane & 15) * ld + kOff + ((lane >> 4) << 3);
  union { uint4 u[2]; bf16x16 v; } c;
  c.u[0] = *(const uint4*)(p);
  c.u[1] = *(const uint4*)(p + 16);
  return c.v;
}

// ---------------------------------------------------------------------------
// One-time prep kernels
// ---------------------------------------------------------------------------
__global__ __launch_bounds__(256) void k_cvt_bf16(const float* __restrict__ src,
                                                  __bf16* __restrict__ dst, int n) {
  for (int i = blockIdx.x * blockDim.x + threadIdx.x; i < n; i += gridDim.x * blockDim.x)
    dst[i] = (__bf16)src[i];
}

// Precompute la_p / la_q (= lrelu(a_shift @ W_act^T + b)) for every (b,t,l), bf16.
__global__ __launch_bounds__(256) void k_la(const float* __restrict__ action,
                                            const float* __restrict__ Wp,
                                            const float* __restrict__ bp,
                                            const float* __restrict__ Wq,
                                            const float* __restrict__ bq,
                                            __bf16* __restrict__ laP,
                                            __bf16* __restrict__ laQ) {
  int idx = blockIdx.x * blockDim.x + threadIdx.x;
  if (idx >= B_ * S_ * L_) return;
  int l  = idx % L_;
  int bt = idx / L_;
  int t  = bt % S_;
  int b  = bt / S_;
  float a0 = 0.f, a1 = 0.f;
  if (t > 0) {
    a0 = action[((size_t)b * S_ + (t - 1)) * A_ + 0];
    a1 = action[((size_t)b * S_ + (t - 1)) * A_ + 1];
  }
  laP[idx] = (__bf16)lrelu(a0 * Wp[l * A_ + 0] + a1 * Wp[l * A_ + 1] + bp[l]);
  laQ[idx] = (__bf16)lrelu(a0 * Wq[l * A_ + 0] + a1 * Wq[l * A_ + 1] + bq[l]);
}

// ---------------------------------------------------------------------------
// Per-step kernels. 64 threads = 2 waves; wave w owns n-tile (2*blockIdx + w)
// and the FULL M=32 (two 16-row A fragments, two accumulators per output).
// ---------------------------------------------------------------------------

// x = lrelu(s_prev @ preW^T + b), emitted as bf16. s_prev = bf16 samp_q buffer.
__global__ __launch_bounds__(64) void k_pre(const __bf16* __restrict__ sPrev,
                                            const __bf16* __restrict__ Wpre,
                                            const float* __restrict__ bias,
                                            __bf16* __restrict__ xOut, int t) {
  int lane  = threadIdx.x & 31;
  int nBase = (blockIdx.x * 2 + (threadIdx.x >> 5)) * 16;
  f32x8 acc[2] = {{}, {}};
  if (t > 0) {
    const __bf16* bRow = Wpre + (size_t)nBase * Z_;
    for (int k = 0; k < Z_; k += 32) {
      bf16x16 b  = b_frag(bRow, Z_, k, lane);
      acc[0] = wmma_bf16(a_frag(sPrev, Z_, k, lane), b, acc[0]);
      acc[1] = wmma_bf16(a_frag(sPrev + 16 * Z_, Z_, k, lane), b, acc[1]);
    }
  }
  int n = nBase + (lane & 15);
  float bn = bias[n];
#pragma unroll
  for (int mt = 0; mt < 2; ++mt)
#pragma unroll
    for (int i = 0; i < 8; ++i) {
      int m = mt * 16 + ((lane >> 4) << 3) + i;
      xOut[(size_t)m * H_ + n] = (__bf16)lrelu(acc[mt][i] + bn);
    }
}

// Fused GRU: gi = x@Wih^T, gh = h@Whh^T, gates, h_new -> fp32 ws + bf16 ws + d_out.
__global__ __launch_bounds__(64) void k_gates(const __bf16* __restrict__ x,
                                              const __bf16* __restrict__ hPrevB,
                                              const float* __restrict__ hPrevF,
                                              const __bf16* __restrict__ Wih,
                                              const __bf16* __restrict__ Whh,
                                              const float* __restrict__ bih,
                                              const float* __restrict__ bhh,
                                              float* __restrict__ hNewF,
                                              __bf16* __restrict__ hNewB,
                                              float* __restrict__ dOut, int t) {
  int lane  = threadIdx.x & 31;
  int nBase = (blockIdx.x * 2 + (threadIdx.x >> 5)) * 16;
  f32x8 iR[2] = {{}, {}}, iZ[2] = {{}, {}}, iN[2] = {{}, {}};
  f32x8 hR[2] = {{}, {}}, hZ[2] = {{}, {}}, hN[2] = {{}, {}};

  {
    const __bf16* wR = Wih + (size_t)(nBase) * H_;
    const __bf16* wZ = Wih + (size_t)(H_ + nBase) * H_;
    const __bf16* wN = Wih + (size_t)(2 * H_ + nBase) * H_;
    for (int k = 0; k < H_; k += 32) {
      bf16x16 a0 = a_frag(x, H_, k, lane);
      bf16x16 a1 = a_frag(x + 16 * H_, H_, k, lane);
      bf16x16 bR = b_frag(wR, H_, k, lane);
      iR[0] = wmma_bf16(a0, bR, iR[0]);
      iR[1] = wmma_bf16(a1, bR, iR[1]);
      bf16x16 bZ = b_frag(wZ, H_, k, lane);
      iZ[0] = wmma_bf16(a0, bZ, iZ[0]);
      iZ[1] = wmma_bf16(a1, bZ, iZ[1]);
      bf16x16 bN = b_frag(wN, H_, k, lane);
      iN[0] = wmma_bf16(a0, bN, iN[0]);
      iN[1] = wmma_bf16(a1, bN, iN[1]);
    }
  }
  if (t > 0) {
    const __bf16* wR = Whh + (size_t)(nBase) * H_;
    const __bf16* wZ = Whh + (size_t)(H_ + nBase) * H_;
    const __bf16* wN = Whh + (size_t)(2 * H_ + nBase) * H_;
    for (int k = 0; k < H_; k += 32) {
      bf16x16 a0 = a_frag(hPrevB, H_, k, lane);
      bf16x16 a1 = a_frag(hPrevB + 16 * H_, H_, k, lane);
      bf16x16 bR = b_frag(wR, H_, k, lane);
      hR[0] = wmma_bf16(a0, bR, hR[0]);
      hR[1] = wmma_bf16(a1, bR, hR[1]);
      bf16x16 bZ = b_frag(wZ, H_, k, lane);
      hZ[0] = wmma_bf16(a0, bZ, hZ[0]);
      hZ[1] = wmma_bf16(a1, bZ, hZ[1]);
      bf16x16 bN = b_frag(wN, H_, k, lane);
      hN[0] = wmma_bf16(a0, bN, hN[0]);
      hN[1] = wmma_bf16(a1, bN, hN[1]);
    }
  }

  int n = nBase + (lane & 15);
  float biR = bih[n], biZ = bih[H_ + n], biN = bih[2 * H_ + n];
  float bhR = bhh[n], bhZ = bhh[H_ + n], bhN = bhh[2 * H_ + n];
#pragma unroll
  for (int mt = 0; mt < 2; ++mt)
#pragma unroll
    for (int i = 0; i < 8; ++i) {
      int m = mt * 16 + ((lane >> 4) << 3) + i;
      float r  = sigm((iR[mt][i] + biR) + (hR[mt][i] + bhR));
      float z  = sigm((iZ[mt][i] + biZ) + (hZ[mt][i] + bhZ));
      float nn = tanh_((iN[mt][i] + biN) + r * (hN[mt][i] + bhN));
      float hp = (t > 0) ? hPrevF[(size_t)m * H_ + n] : 0.f;
      float h  = (1.f - z) * nn + z * hp;
      hNewF[(size_t)m * H_ + n] = h;
      hNewB[(size_t)m * H_ + n] = (__bf16)h;
      size_t idx = ((size_t)m * S_ + t) * H_ + n;
      dOut[O0 + idx] = h;
      dOut[O4 + idx] = h;
    }
}

// Hidden layers of both MLPs. n-tiles [0,68): prior (K=1088); [68,200): post
// (K=2112). Concat handled by K-tile source switching: h | embedding | la.
__global__ __launch_bounds__(64) void k_mid(const __bf16* __restrict__ hCur,
                                            const __bf16* __restrict__ embB,
                                            const __bf16* __restrict__ laP,
                                            const __bf16* __restrict__ laQ,
                                            const __bf16* __restrict__ Wp1,
                                            const float* __restrict__ bp1,
                                            const __bf16* __restrict__ Wq1,
                                            const float* __restrict__ bq1,
                                            __bf16* __restrict__ hmP,
                                            __bf16* __restrict__ hmQ, int t) {
  int lane  = threadIdx.x & 31;
  int tile  = blockIdx.x * 2 + (threadIdx.x >> 5);
  bool isQ  = tile >= (CP_ / 16);
  int nBase = (isQ ? tile - CP_ / 16 : tile) * 16;
  int K     = isQ ? CQ_ : CP_;
  const __bf16* W   = (isQ ? Wq1 : Wp1) + (size_t)nBase * K;
  const float* bias = isQ ? bq1 : bp1;
  const __bf16* la  = isQ ? laQ : laP;
  __bf16* outBuf    = isQ ? hmQ : hmP;

  f32x8 acc[2] = {{}, {}};
  for (int k = 0; k < K; k += 32) {
    const __bf16* aBase; size_t aStride; int kOff;
    if (k < H_) {                                   // h_new part
      aBase = hCur;                                 aStride = H_;             kOff = k;
    } else if (isQ && k < H_ + E_) {                // embedding part (post only)
      aBase = embB + (size_t)t * E_;                aStride = (size_t)S_ * E_; kOff = k - H_;
    } else {                                        // la part
      int laStart = isQ ? (H_ + E_) : H_;
      aBase = la + (size_t)t * L_;                  aStride = (size_t)S_ * L_; kOff = k - laStart;
    }
    bf16x16 b = b_frag(W, K, k, lane);
    acc[0] = wmma_bf16(a_frag(aBase, aStride, kOff, lane), b, acc[0]);
    acc[1] = wmma_bf16(a_frag(aBase + 16 * aStride, aStride, kOff, lane), b, acc[1]);
  }

  int n = nBase + (lane & 15);
  float bn = bias[n];
#pragma unroll
  for (int mt = 0; mt < 2; ++mt)
#pragma unroll
    for (int i = 0; i < 8; ++i) {
      int m = mt * 16 + ((lane >> 4) << 3) + i;
      outBuf[(size_t)m * K + n] = (__bf16)lrelu(acc[mt][i] + bn);
    }
}

// Output layers + sampling. n-tiles [0,32): prior; [32,64): post. Each wave
// computes a mu tile AND its paired log-sigma tile (cols 512+n) for M=32, so
// sig/samp fuse in-register. Post branch also emits bf16 samp_q for next step.
__global__ __launch_bounds__(64) void k_out(const __bf16* __restrict__ hmP,
                                            const __bf16* __restrict__ hmQ,
                                            const __bf16* __restrict__ Wp2,
                                            const float* __restrict__ bp2,
                                            const __bf16* __restrict__ Wq2,
                                            const float* __restrict__ bq2,
                                            const float* __restrict__ nzP,
                                            const float* __restrict__ nzQ,
                                            __bf16* __restrict__ sampQB,
                                            float* __restrict__ dOut, int t) {
  int lane = threadIdx.x & 31;
  int tile = blockIdx.x * 2 + (threadIdx.x >> 5);
  bool isQ = tile >= (Z_ / 16);
  int nMu  = (isQ ? tile - Z_ / 16 : tile) * 16;
  int K    = isQ ? CQ_ : CP_;
  const __bf16* hm   = isQ ? hmQ : hmP;
  const __bf16* W    = isQ ? Wq2 : Wp2;
  const float*  bias = isQ ? bq2 : bp2;
  const float*  nz   = isQ ? nzQ : nzP;
  const __bf16* Wmu  = W + (size_t)nMu * K;
  const __bf16* Wls  = W + (size_t)(Z_ + nMu) * K;

  f32x8 aMu[2] = {{}, {}}, aLs[2] = {{}, {}};
  for (int k = 0; k < K; k += 32) {
    bf16x16 a0 = a_frag(hm, K, k, lane);
    bf16x16 a1 = a_frag(hm + 16 * K, K, k, lane);
    bf16x16 bM = b_frag(Wmu, K, k, lane);
    aMu[0] = wmma_bf16(a0, bM, aMu[0]);
    aMu[1] = wmma_bf16(a1, bM, aMu[1]);
    bf16x16 bL = b_frag(Wls, K, k, lane);
    aLs[0] = wmma_bf16(a0, bL, aLs[0]);
    aLs[1] = wmma_bf16(a1, bL, aLs[1]);
  }

  int zc = nMu + (lane & 15);
  float bM = bias[zc], bL = bias[Z_ + zc];
  size_t oMu = isQ ? O6 : O2, oSg = isQ ? O7 : O3, oSa = isQ ? O5 : O1;
#pragma unroll
  for (int mt = 0; mt < 2; ++mt)
#pragma unroll
    for (int i = 0; i < 8; ++i) {
      int m = mt * 16 + ((lane >> 4) << 3) + i;
      size_t idx = ((size_t)m * S_ + t) * Z_ + zc;
      float mu = aMu[mt][i] + bM;
      float sg = 2.f * sigm(0.5f * (aLs[mt][i] + bL)) + 0.1f;
      float sa = mu + sg * nz[idx];
      dOut[oMu + idx] = mu;
      dOut[oSg + idx] = sg;
      dOut[oSa + idx] = sa;
      if (isQ) sampQB[(size_t)m * Z_ + zc] = (__bf16)sa;  // feeds next k_pre
    }
}

// ---------------------------------------------------------------------------
// Host
// ---------------------------------------------------------------------------
extern "C" void kernel_launch(void* const* d_in, const int* in_sizes, int n_in,
                              void* d_out, int out_size, void* d_ws, size_t ws_size,
                              hipStream_t stream) {
  const float* emb  = (const float*)d_in[0];
  const float* act  = (const float*)d_in[1];
  const float* nzP  = (const float*)d_in[2];
  const float* nzQ  = (const float*)d_in[3];
  const float* preW = (const float*)d_in[4];
  const float* preB = (const float*)d_in[5];
  const float* Wih  = (const float*)d_in[6];
  const float* Whh  = (const float*)d_in[7];
  const float* bih  = (const float*)d_in[8];
  const float* bhh  = (const float*)d_in[9];
  const float* paW  = (const float*)d_in[10];
  const float* paB  = (const float*)d_in[11];
  const float* prW  = (const float*)d_in[12];
  const float* prB  = (const float*)d_in[13];
  const float* pW1  = (const float*)d_in[14];
  const float* pB1  = (const float*)d_in[15];
  const float* pW2  = (const float*)d_in[16];
  const float* pB2  = (const float*)d_in[17];
  const float* qW1  = (const float*)d_in[18];  // prior_W1
  const float* qB1  = (const float*)d_in[19];  // prior_b1
  const float* qW2  = (const float*)d_in[20];  // prior_W2
  const float* qB2  = (const float*)d_in[21];  // prior_b2

  char* w = (char*)d_ws;
  auto take = [&](size_t bytes) {
    void* p = (void*)w;
    w += (bytes + 255) & ~(size_t)255;
    return p;
  };
  __bf16* bwPre = (__bf16*)take(2ull * H_ * Z_);
  __bf16* bwIh  = (__bf16*)take(2ull * 3 * H_ * H_);
  __bf16* bwHh  = (__bf16*)take(2ull * 3 * H_ * H_);
  __bf16* bwQ1  = (__bf16*)take(2ull * CQ_ * CQ_);     // post_W1
  __bf16* bwQ2  = (__bf16*)take(2ull * 2 * Z_ * CQ_);  // post_W2
  __bf16* bwP1  = (__bf16*)take(2ull * CP_ * CP_);     // prior_W1
  __bf16* bwP2  = (__bf16*)take(2ull * 2 * Z_ * CP_);  // prior_W2
  __bf16* embB  = (__bf16*)take(2ull * B_ * S_ * E_);  // bf16 embeddings
  __bf16* laP   = (__bf16*)take(2ull * B_ * S_ * L_);
  __bf16* laQ   = (__bf16*)take(2ull * B_ * S_ * L_);
  __bf16* xB    = (__bf16*)take(2ull * B_ * H_);
  float*  hF    = (float*)take(4ull * 2 * B_ * H_);    // fp32 h ping-pong
  __bf16* hB    = (__bf16*)take(2ull * 2 * B_ * H_);   // bf16 h ping-pong
  __bf16* hmP   = (__bf16*)take(2ull * B_ * CP_);
  __bf16* hmQ   = (__bf16*)take(2ull * B_ * CQ_);
  __bf16* sB    = (__bf16*)take(2ull * B_ * Z_);       // bf16 samp_q

  auto cvt = [&](const float* s, __bf16* d, size_t n) {
    int blocks = (int)((n + 255) / 256);
    if (blocks > 16384) blocks = 16384;
    k_cvt_bf16<<<blocks, 256, 0, stream>>>(s, d, (int)n);
  };
  cvt(preW, bwPre, (size_t)H_ * Z_);
  cvt(Wih,  bwIh,  (size_t)3 * H_ * H_);
  cvt(Whh,  bwHh,  (size_t)3 * H_ * H_);
  cvt(pW1,  bwQ1,  (size_t)CQ_ * CQ_);
  cvt(pW2,  bwQ2,  (size_t)2 * Z_ * CQ_);
  cvt(qW1,  bwP1,  (size_t)CP_ * CP_);
  cvt(qW2,  bwP2,  (size_t)2 * Z_ * CP_);
  cvt(emb,  embB,  (size_t)B_ * S_ * E_);

  k_la<<<(B_ * S_ * L_ + 255) / 256, 256, 0, stream>>>(act, prW, prB, paW, paB, laP, laQ);

  float* out = (float*)d_out;
  for (int t = 0; t < S_; ++t) {
    float*  hCurF = hF + (size_t)(t & 1) * B_ * H_;
    __bf16* hCurB = hB + (size_t)(t & 1) * B_ * H_;
    const float*  hPrevF = hF + (size_t)((t + 1) & 1) * B_ * H_;
    const __bf16* hPrevB = hB + (size_t)((t + 1) & 1) * B_ * H_;

    k_pre  <<<H_ / 32, 64, 0, stream>>>(sB, bwPre, preB, xB, t);
    k_gates<<<H_ / 32, 64, 0, stream>>>(xB, hPrevB, hPrevF, bwIh, bwHh, bih, bhh,
                                        hCurF, hCurB, out, t);
    k_mid  <<<(CP_ / 16 + CQ_ / 16) / 2, 64, 0, stream>>>(hCurB, embB, laP, laQ,
                                                          bwP1, qB1, bwQ1, pB1,
                                                          hmP, hmQ, t);
    k_out  <<<Z_ / 16, 64, 0, stream>>>(hmP, hmQ, bwP2, qB2, bwQ2, pB2,
                                        nzP, nzQ, sB, out, t);
  }
}
